// ScannedLSTM_61186104099127
// MI455X (gfx1250) — matrix operand order repaired and checked by reading, not compiled, not run on hardware
//
#include <hip/hip_runtime.h>
#include <stdint.h>

// ---------------------------------------------------------------------------
// Scanned LSTM for MI455X (gfx1250, wave32, WMMA).
// T=512, B=128, D=512.  z = xt@Wi + h@Wh + b ; gates -> c,h update.
// bf16 WMMA (16x16x32, f32 accum); weights pre-transposed to bf16 once
// (4MB, L2-resident); one fused kernel launch per timestep.
// Each wave: M=32 x N=16 output patch, all 4 gates fused -> every B tile
// feeds 2 WMMAs, every A tile feeds 4; 16 back-to-back WMMAs per k-slice.
// ---------------------------------------------------------------------------

typedef __attribute__((ext_vector_type(16))) __bf16 v16bf;
typedef __attribute__((ext_vector_type(8)))  float  v8f;
typedef __attribute__((ext_vector_type(4)))  float  v4f;

#define T_STEPS 512
#define BATCH   128
#define DIM     512
#define G4      2048   // 4*DIM

// ---------------------------------------------------------------------------
// Transpose + convert W (DIM x G4, f32, row-major) -> WT (G4 x DIM, bf16,
// row-major).  WT row n = K-contiguous column n of W: this kernel's WMMA
// B-operand feed (lane = N, contiguous K per lane, lanes 16-31 at K+16).
// ---------------------------------------------------------------------------
__global__ void wt_convert_kernel(const float* __restrict__ W,
                                  __bf16* __restrict__ WT) {
    int k = blockIdx.x * 64 + threadIdx.x;   // 0..DIM-1
    int n = blockIdx.y;                      // 0..G4-1
    WT[(size_t)n * DIM + k] = (__bf16)W[(size_t)k * G4 + n];
}

__global__ void c_init_kernel(const float* __restrict__ c0,
                              float* __restrict__ c_state) {
    int i = blockIdx.x * 256 + threadIdx.x;
    c_state[i] = c0[i];
}

// ---------------------------------------------------------------------------
// A-operand loader: 16x32 bf16 tile row from f32 (row stride DIM).
// A layout used here (contiguous K, same as B): lane L holds row M=L%16;
//   vec[0..15] = K = (L/16)*16 + 0..15
// msk = 0 zeroes the whole row (dones reset of h) with 8 v_and_b32 on the
// packed bf16 pairs -- branchless, EXEC stays all-ones for WMMA.
// ---------------------------------------------------------------------------
__device__ inline v16bf load_a_f32(const float* __restrict__ rowk, int half,
                                   uint32_t msk) {
    const float* p = rowk + half * 16;
    v4f f0 = *(const v4f*)(p);
    v4f f1 = *(const v4f*)(p + 4);
    v4f f2 = *(const v4f*)(p + 8);
    v4f f3 = *(const v4f*)(p + 12);
    union { v16bf v; uint32_t u[8]; } r;
#pragma unroll
    for (int i = 0; i < 4; ++i) {
        r.v[i]      = (__bf16)f0[i];
        r.v[4 + i]  = (__bf16)f1[i];
        r.v[8 + i]  = (__bf16)f2[i];
        r.v[12 + i] = (__bf16)f3[i];
    }
#pragma unroll
    for (int i = 0; i < 8; ++i) r.u[i] &= msk;
    return r.v;
}

__device__ inline v8f wmma_bf16(v16bf a, v16bf b, v8f c) {
    return __builtin_amdgcn_wmma_f32_16x16x32_bf16(
        false, a, false, b, (short)0, c, false, false);
}

__device__ inline float sigmoid_f(float x) { return 1.0f / (1.0f + __expf(-x)); }

// ---------------------------------------------------------------------------
// One LSTM timestep.  One wave per 32x16 (batch x dim) output patch; all 4
// gates fused.  grid = (BATCH/32)*(DIM/16) = 128 blocks, blockDim = 32.
// ---------------------------------------------------------------------------
__global__ __launch_bounds__(32)
void lstm_step_kernel(const float* __restrict__ xt,          // (B, D) f32
                      const float* __restrict__ hprev,       // (B, D) f32
                      const unsigned char* __restrict__ dn,  // (B,) bool8
                      const __bf16* __restrict__ WiT,        // (G4, D) bf16
                      const __bf16* __restrict__ WhT,        // (G4, D) bf16
                      const float* __restrict__ bias,        // (G4,) f32
                      float* __restrict__ c_state,           // (B, D) f32
                      float* __restrict__ h_out)             // ys[t] (B, D)
{
    const int lane = threadIdx.x & 31;
    const int half = lane >> 4;        // 0 or 1
    const int lrow = lane & 15;

    const int tile = blockIdx.x;       // 0..127
    const int m0 = (tile & 3) * 32;    // batch patch base (two 16-row tiles)
    const int n0 = (tile >> 2) * 16;   // dim tile base
    const int n  = n0 + lrow;

    // f32 accumulators [gate][m-tile], initialized with the bias (uniform
    // over rows of the C tile, so all 8 elements get bias[g*512+n]).
    v8f acc[4][2];
#pragma unroll
    for (int g = 0; g < 4; ++g) {
        const float bg = bias[g * 512 + n];
#pragma unroll
        for (int mt = 0; mt < 2; ++mt)
#pragma unroll
            for (int j = 0; j < 8; ++j) acc[g][mt][j] = bg;
    }

    // A-operand rows for this lane (one batch row per lane per m-tile)
    const float* xrow0 = xt + (size_t)(m0 + lrow) * DIM;
    const float* xrow1 = xrow0 + 16 * DIM;
    const float* hrow0 = hprev + (size_t)(m0 + lrow) * DIM;
    const float* hrow1 = hrow0 + 16 * DIM;
    const uint32_t hmsk0 = dn[m0 + lrow] ? 0u : 0xffffffffu;
    const uint32_t hmsk1 = dn[m0 + 16 + lrow] ? 0u : 0xffffffffu;

    // B-operand base: WT row = output column (gate*512 + n); lanes 0-15
    // feed K=0..15, lanes 16-31 feed K=16..31 (contiguous-K order, same as A)
    const size_t colbase = (size_t)n * DIM + (size_t)half * 16;

#pragma unroll 2
    for (int k0 = 0; k0 < DIM; k0 += 32) {
        // ---- issue ALL operand loads for this k-slice first (one clause,
        // ---- one wait), then 16 back-to-back WMMAs --------------------
        v16bf bi[4], bh[4];
#pragma unroll
        for (int g = 0; g < 4; ++g) {
            const size_t goff = (size_t)g * (512 * DIM) + colbase + k0;
            bi[g] = *(const v16bf*)(WiT + goff);
            bh[g] = *(const v16bf*)(WhT + goff);
        }
        v16bf ax0 = load_a_f32(xrow0 + k0, half, 0xffffffffu);
        v16bf ax1 = load_a_f32(xrow1 + k0, half, 0xffffffffu);
        v16bf ah0 = load_a_f32(hrow0 + k0, half, hmsk0);
        v16bf ah1 = load_a_f32(hrow1 + k0, half, hmsk1);

#pragma unroll
        for (int g = 0; g < 4; ++g) {
            acc[g][0] = wmma_bf16(ax0, bi[g], acc[g][0]);
            acc[g][1] = wmma_bf16(ax1, bi[g], acc[g][1]);
        }
#pragma unroll
        for (int g = 0; g < 4; ++g) {
            acc[g][0] = wmma_bf16(ah0, bh[g], acc[g][0]);
            acc[g][1] = wmma_bf16(ah1, bh[g], acc[g][1]);
        }
    }

    // ---- fused gate math + state update -----------------------------------
    // C/D layout: acc[..][mt][j] <-> row M = m0 + mt*16 + half*8 + j, col n
#pragma unroll
    for (int mt = 0; mt < 2; ++mt) {
        const int mbase = m0 + mt * 16 + half * 8;
        const uint64_t flags8 = *(const uint64_t*)(dn + mbase);
#pragma unroll
        for (int j = 0; j < 8; ++j) {
            const bool reset = ((flags8 >> (8 * j)) & 0xffull) != 0;

            const float ig = sigmoid_f(acc[0][mt][j]);
            const float fg = sigmoid_f(acc[1][mt][j]);
            const float gg = tanhf(acc[2][mt][j]);
            const float og = sigmoid_f(acc[3][mt][j]);

            const size_t idx = (size_t)(mbase + j) * DIM + n;
            const float c_old = reset ? 0.0f : c_state[idx];
            const float c_new = fg * c_old + ig * gg;
            c_state[idx] = c_new;
            h_out[idx] = og * tanhf(c_new);
        }
    }
}

// ---------------------------------------------------------------------------
extern "C" void kernel_launch(void* const* d_in, const int* in_sizes, int n_in,
                              void* d_out, int out_size, void* d_ws, size_t ws_size,
                              hipStream_t stream) {
    const float* x          = (const float*)d_in[0];          // (T,B,D)
    const unsigned char* dn = (const unsigned char*)d_in[1];  // (T,B) bool8
    const float* c0         = (const float*)d_in[2];          // (B,D)
    const float* h0         = (const float*)d_in[3];          // (B,D)
    const float* Wi         = (const float*)d_in[4];          // (D,4D)
    const float* Wh         = (const float*)d_in[5];          // (D,4D)
    const float* bias       = (const float*)d_in[6];          // (4D,)
    float* ys = (float*)d_out;                                 // (T,B,D)

    // workspace layout (~4.5 MB): WiT(2MB) | WhT(2MB) | c_state(256KB)
    char* ws = (char*)d_ws;
    __bf16* WiT  = (__bf16*)ws;
    __bf16* WhT  = (__bf16*)(ws + (size_t)G4 * DIM * sizeof(__bf16));
    float*  c_st = (float*) (ws + (size_t)2 * G4 * DIM * sizeof(__bf16));

    // one-time (per call) weight convert+transpose and c-state init
    dim3 cg(DIM / 64, G4);
    wt_convert_kernel<<<cg, 64, 0, stream>>>(Wi, WiT);
    wt_convert_kernel<<<cg, 64, 0, stream>>>(Wh, WhT);
    c_init_kernel<<<(BATCH * DIM) / 256, 256, 0, stream>>>(c0, c_st);

    // sequential scan: ys[t-1] serves as h input of step t (stream-ordered)
    for (int t = 0; t < T_STEPS; ++t) {
        const float* xt = x + (size_t)t * BATCH * DIM;
        const float* hp = (t == 0) ? h0 : (ys + (size_t)(t - 1) * BATCH * DIM);
        lstm_step_kernel<<<128, 32, 0, stream>>>(
            xt, hp, dn + (size_t)t * BATCH, WiT, WhT, bias, c_st,
            ys + (size_t)t * BATCH * DIM);
    }
}
